// VoxelGNNDiscriminator_66546223284348
// MI455X (gfx1250) — matrix operand order repaired and checked by reading, not compiled, not run on hardware
//
#include <hip/hip_runtime.h>
#include <hip/hip_bf16.h>

#define N_LOCALC 400
#define N_VOXELC 200000
#define N_EDGESC 1200000
#define D_LOCALC 19
#define D_VOXELC 6
#define NUM_CLS 7
#define GN_EPS 1e-5f

typedef __attribute__((ext_vector_type(16))) _Float16 v16h;
typedef __attribute__((ext_vector_type(8)))  _Float16 v8h;
typedef __attribute__((ext_vector_type(8)))  float    v8f;

__device__ __forceinline__ void atomAddF(float* p, float v) {
  unsafeAtomicAdd(p, v);   // global_atomic_add_f32 on gfx1250
}

// ---------------------------------------------------------------- class means
__global__ void k_class_means(const float* __restrict__ lx,
                              const int* __restrict__ lt,
                              float* __restrict__ means) {
  int t = threadIdx.x;
  if (t >= NUM_CLS * D_LOCALC) return;
  int c = t / D_LOCALC, d = t % D_LOCALC;
  float s = 0.f; int cnt = 0;
  for (int i = 0; i < N_LOCALC; ++i) {
    if (lt[i] == c) { s += lx[i * D_LOCALC + d]; ++cnt; }
  }
  means[t] = (cnt > 0) ? s / (float)cnt : 0.0f;
}

// --------------------------------- build x0 = concat features (f16 activation)
__global__ void k_build_x0(const float* __restrict__ means,
                           const int* __restrict__ vtype,
                           const float* __restrict__ vx,
                           const float* __restrict__ lab,
                           _Float16* __restrict__ x0) {
  int i = blockIdx.x * blockDim.x + threadIdx.x;
  if (i >= N_VOXELC) return;
  const float* m = means + vtype[i] * D_LOCALC;
  _Float16* o = x0 + (size_t)i * 32;
#pragma unroll
  for (int d = 0; d < D_LOCALC; ++d) o[d] = (_Float16)m[d];
  const float* v = vx + (size_t)i * D_VOXELC;
#pragma unroll
  for (int d = 0; d < D_VOXELC; ++d) o[D_LOCALC + d] = (_Float16)v[d];
  const float* l = lab + (size_t)i * NUM_CLS;
#pragma unroll
  for (int d = 0; d < NUM_CLS; ++d) o[D_LOCALC + D_VOXELC + d] = (_Float16)l[d];
}

// ------------------------------------- pack fp32 weights into WMMA B fragments
// B layout per ISA 7.12: lane%16 = N col, 16 halves = K rows (8h+s / 16+8h+s-8), h = lane/16.
__global__ void k_pack_w(const float* __restrict__ W, _Float16* __restrict__ dst,
                         int K, int N) {
  int tid = blockIdx.x * blockDim.x + threadIdx.x;
  if (tid >= K * N) return;
  int j    = tid & 15;
  int lane = (tid >> 4) & 31;
  int tile = tid >> 9;            // 32 lanes * 16 halves = 512 per tile
  int Ntiles = N >> 4;
  int nt = tile % Ntiles, kt = tile / Ntiles;
  int h = lane >> 4;
  int k = kt * 32 + (j < 8 ? 8 * h + j : 16 + 8 * h + (j - 8));
  int n = nt * 16 + (lane & 15);
  dst[tid] = (_Float16)W[(size_t)k * N + n];
}

// ------------------------------------------------------ WMMA GEMM (M=200000)
// A is f16 [M,K]; Wp pre-packed f16; one wave -> 16 x (TN*16) output tile.
// FUSE: add bias + relu.  OUT16: write f16 activations, else f32.
template<int TN, int FUSE, int OUT16>
__global__ __launch_bounds__(256)
void k_gemm(const _Float16* __restrict__ X, const _Float16* __restrict__ Wp,
            const float* __restrict__ bias, void* __restrict__ Yv,
            int K, int N) {
  int wid  = (blockIdx.x * blockDim.x + threadIdx.x) >> 5;
  int lane = threadIdx.x & 31;
  int Ntiles = N >> 4;
  int groups = Ntiles / TN;
  int Mtiles = N_VOXELC >> 4;
  if (wid >= Mtiles * groups) return;
  int mt = wid / groups;
  int ng = wid % groups;
  int ml = lane & 15;
  int g  = lane >> 4;
  int row = mt * 16 + ml;
  const _Float16* xrow = X + (size_t)row * K + 8 * g;

  v8f acc[TN] = {};
  int Kt = K >> 5;
  for (int kt = 0; kt < Kt; ++kt) {
    // A fragment (ISA layout): lane holds X[row][kt*32+8g..+7] and [kt*32+16+8g..+7]
    v8h alo = *(const v8h*)(xrow + kt * 32);
    v8h ahi = *(const v8h*)(xrow + kt * 32 + 16);
    v16h a;
#pragma unroll
    for (int j = 0; j < 8; ++j) { a[j] = alo[j]; a[8 + j] = ahi[j]; }
    const _Float16* bbase = Wp + (((size_t)kt * Ntiles + ng * TN) * 32 + lane) * 16;
#pragma unroll
    for (int t = 0; t < TN; ++t) {
      v16h b = *(const v16h*)(bbase + (size_t)t * 512);
      acc[t] = __builtin_amdgcn_wmma_f32_16x16x32_f16(
          false, a, false, b, (short)0, acc[t], false, false);
    }
  }
  // C layout: lane n = lane&15 ; VGPR r -> m = r + 8*(lane>>4)
  int mbase = mt * 16 + 8 * g;
#pragma unroll
  for (int t = 0; t < TN; ++t) {
    int n = (ng * TN + t) * 16 + ml;
    float bv = FUSE ? bias[n] : 0.0f;
#pragma unroll
    for (int r = 0; r < 8; ++r) {
      float v = acc[t][r] + bv;
      if (FUSE) v = fmaxf(v, 0.0f);
      if (OUT16) ((_Float16*)Yv)[(size_t)(mbase + r) * N + n] = (_Float16)v;
      else       ((float*)   Yv)[(size_t)(mbase + r) * N + n] = v;
    }
  }
}

// ------------------------------------------------------------- degree / dinv
__global__ void k_deg_init(float* __restrict__ deg) {
  int i = blockIdx.x * blockDim.x + threadIdx.x;
  if (i < N_VOXELC) deg[i] = 1.0f;           // self loop
}
__global__ void k_deg_count(const int* __restrict__ dst, float* __restrict__ deg) {
  int e = blockIdx.x * blockDim.x + threadIdx.x;
  if (e < N_EDGESC) atomAddF(&deg[dst[e]], 1.0f);
}
__global__ void k_dinv(float* __restrict__ deg) {
  int i = blockIdx.x * blockDim.x + threadIdx.x;
  if (i < N_VOXELC) deg[i] = rsqrtf(fmaxf(deg[i], 1.0f));
}

// ------------------------------- GCN aggregation (self + edges), float4 wide
// qshift: log2(N/4)
__global__ void k_self_init4(const float* __restrict__ xw,
                             const float* __restrict__ dinv,
                             const float* __restrict__ bias,
                             float* __restrict__ out, int qshift) {
  size_t i = (size_t)blockIdx.x * blockDim.x + threadIdx.x;
  if (i >= ((size_t)N_VOXELC << qshift)) return;
  int node = (int)(i >> qshift);
  int q    = (int)(i & ((1u << qshift) - 1));
  float di = dinv[node];
  float s  = di * di;
  float4 v = ((const float4*)xw)[i];
  float4 b = ((const float4*)bias)[q];
  float4 o; o.x = v.x*s + b.x; o.y = v.y*s + b.y; o.z = v.z*s + b.z; o.w = v.w*s + b.w;
  ((float4*)out)[i] = o;
}
__global__ void k_edge_scatter4(const int* __restrict__ src,
                                const int* __restrict__ dst,
                                const float* __restrict__ xw,
                                const float* __restrict__ dinv,
                                float* __restrict__ out, int qshift) {
  size_t tid = (size_t)blockIdx.x * blockDim.x + threadIdx.x;
  if (tid >= ((size_t)N_EDGESC << qshift)) return;
  int e = (int)(tid >> qshift);
  int q = (int)(tid & ((1u << qshift) - 1));
  int s = src[e], d = dst[e];
  float sc = dinv[s] * dinv[d];
  float4 v = ((const float4*)xw)[((size_t)s << qshift) + q];
  float* o = out + ((((size_t)d << qshift) + q) << 2);
  atomAddF(o + 0, v.x * sc);
  atomAddF(o + 1, v.y * sc);
  atomAddF(o + 2, v.z * sc);
  atomAddF(o + 3, v.w * sc);
}

// ----------------------------------------------------------------- GraphNorm
__global__ void k_zero(float* __restrict__ p, int n) {
  int i = blockIdx.x * blockDim.x + threadIdx.x;
  if (i < n) p[i] = 0.0f;
}
#define GN_CHUNK 512
__global__ void k_gn_stats(const float* __restrict__ x, float* __restrict__ stats,
                           int N) {
  int f  = threadIdx.x;                 // blockDim.x == N
  int r0 = blockIdx.x * GN_CHUNK;
  int r1 = min(r0 + GN_CHUNK, N_VOXELC);
  float s = 0.f, s2 = 0.f;
  for (int r = r0; r < r1; ++r) {
    float v = x[(size_t)r * N + f];
    s += v; s2 += v * v;
  }
  atomAddF(&stats[f], s);
  atomAddF(&stats[N + f], s2);
}
// norm + relu, f32 in -> f16 activation out (producer-side cvt for next GEMM)
__global__ void k_gn_apply(const float* __restrict__ x, const float* __restrict__ stats,
                           const float* __restrict__ gw, const float* __restrict__ gb,
                           const float* __restrict__ gm, _Float16* __restrict__ y,
                           int nshift) {
  size_t i = (size_t)blockIdx.x * blockDim.x + threadIdx.x;
  if (i >= ((size_t)N_VOXELC << nshift)) return;
  int N = 1 << nshift;
  int f = (int)(i & (N - 1));
  const float invM = 1.0f / (float)N_VOXELC;
  float m   = stats[f] * invM;
  float ex2 = stats[N + f] * invM;
  float ms  = gm[f];
  float sm  = ms * m;
  float var = ex2 - 2.0f * sm * m + sm * sm;   // E[(x - ms*m)^2]
  float v = (x[i] - sm) * rsqrtf(var + GN_EPS);
  y[i] = (_Float16)fmaxf(gw[f] * v + gb[f], 0.0f);
}

// ------------------------------------------------------ final 16->1 + sigmoid
__global__ void k_final(const _Float16* __restrict__ x, const float* __restrict__ W,
                        const float* __restrict__ b, float* __restrict__ out) {
  int i = blockIdx.x * blockDim.x + threadIdx.x;
  if (i >= N_VOXELC) return;
  float s = b[0];
#pragma unroll
  for (int d = 0; d < 16; ++d) s += (float)x[(size_t)i * 16 + d] * W[d];
  out[i] = 1.0f / (1.0f + expf(-s));
}

// =============================================================== host driver
extern "C" void kernel_launch(void* const* d_in, const int* in_sizes, int n_in,
                              void* d_out, int out_size, void* d_ws, size_t ws_size,
                              hipStream_t stream) {
  const float* local_x   = (const float*)d_in[0];
  const int*   local_t   = (const int*)  d_in[1];
  const float* voxel_x   = (const float*)d_in[2];
  const int*   voxel_t   = (const int*)  d_in[3];
  const int*   edge      = (const int*)  d_in[4];
  const float* label     = (const float*)d_in[5];
  const float* We1 = (const float*)d_in[6];  const float* be1 = (const float*)d_in[7];
  const float* We2 = (const float*)d_in[8];  const float* be2 = (const float*)d_in[9];
  const float *Wc[4], *bc[4], *gw[4], *gb[4], *gm[4];
  for (int i = 0; i < 4; ++i) {
    Wc[i] = (const float*)d_in[10 + 5 * i + 0];
    bc[i] = (const float*)d_in[10 + 5 * i + 1];
    gw[i] = (const float*)d_in[10 + 5 * i + 2];
    gb[i] = (const float*)d_in[10 + 5 * i + 3];
    gm[i] = (const float*)d_in[10 + 5 * i + 4];
  }
  const float* Wd0 = (const float*)d_in[30]; const float* bd0 = (const float*)d_in[31];
  const float* Wd1 = (const float*)d_in[32]; const float* bd1 = (const float*)d_in[33];
  const float* Wd2 = (const float*)d_in[34]; const float* bd2 = (const float*)d_in[35];
  const float* Wd3 = (const float*)d_in[36]; const float* bd3 = (const float*)d_in[37];
  const int* srcp = edge;
  const int* dstp = edge + N_EDGESC;
  float* out = (float*)d_out;

  // workspace layout
  const size_t SF = (size_t)N_VOXELC * 128 * sizeof(_Float16); // 51.2 MB
  const size_t SG = (size_t)N_VOXELC * 128 * sizeof(float);    // 102.4 MB
  char* ws = (char*)d_ws;
  _Float16* F0 = (_Float16*)(ws);                 // f16 activations (ping)
  _Float16* F1 = (_Float16*)(ws + SF);            // f16 activations (pong)
  float* G0    = (float*)(ws + 2 * SF);           // f32 xw
  float* G1    = (float*)(ws + 2 * SF + SG);      // f32 agg
  float* dinv  = (float*)(ws + 2 * SF + 2 * SG);  // 800 KB
  float* stats = (float*)(ws + 2 * SF + 2 * SG + (1 << 20));
  float* cmean = (float*)(ws + 2 * SF + 2 * SG + (1 << 20) + 4096);
  _Float16* pw = (_Float16*)(ws + 2 * SF + 2 * SG + (1 << 20) + 8192);

  // packed-weight offsets (halves)
  const size_t oWe1 = 0,      oWe2 = 4096,  oWc0 = 20480, oWc1 = 28672;
  const size_t oWc2 = 30720,  oWc3 = 32768, oWd0 = 40960, oWd1 = 49152, oWd2 = 51200;

  auto pack = [&](const float* W, size_t off, int K, int N) {
    int tot = K * N;
    k_pack_w<<<(tot + 255) / 256, 256, 0, stream>>>(W, pw + off, K, N);
  };
  pack(We1, oWe1, 32, 128);  pack(We2, oWe2, 128, 128);
  pack(Wc[0], oWc0, 128, 64); pack(Wc[1], oWc1, 64, 32);
  pack(Wc[2], oWc2, 32, 64);  pack(Wc[3], oWc3, 64, 128);
  pack(Wd0, oWd0, 128, 64);   pack(Wd1, oWd1, 64, 32);  pack(Wd2, oWd2, 32, 16);

  auto gemm = [&](const _Float16* X, size_t woff, const float* bias, void* Y,
                  int K, int N, bool fuse16) {
    int Ntiles = N / 16;
    int TN = Ntiles >= 4 ? 4 : Ntiles;
    int groups = Ntiles / TN;
    int waves = (N_VOXELC / 16) * groups;
    int blocks = (waves * 32 + 255) / 256;
    const _Float16* Wp = pw + woff;
    if (fuse16) {          // bias + relu, f16 activation out
      if (TN == 4)      k_gemm<4,1,1><<<blocks,256,0,stream>>>(X, Wp, bias, Y, K, N);
      else if (TN == 2) k_gemm<2,1,1><<<blocks,256,0,stream>>>(X, Wp, bias, Y, K, N);
      else              k_gemm<1,1,1><<<blocks,256,0,stream>>>(X, Wp, bias, Y, K, N);
    } else {               // raw xw, f32 out
      if (TN == 4)      k_gemm<4,0,0><<<blocks,256,0,stream>>>(X, Wp, bias, Y, K, N);
      else if (TN == 2) k_gemm<2,0,0><<<blocks,256,0,stream>>>(X, Wp, bias, Y, K, N);
      else              k_gemm<1,0,0><<<blocks,256,0,stream>>>(X, Wp, bias, Y, K, N);
    }
  };

  // --- preamble: class means, features, degrees
  k_class_means<<<1, 160, 0, stream>>>(local_x, local_t, cmean);
  k_build_x0<<<(N_VOXELC + 255) / 256, 256, 0, stream>>>(cmean, voxel_t, voxel_x, label, F0);
  k_deg_init<<<(N_VOXELC + 255) / 256, 256, 0, stream>>>(dinv);
  k_deg_count<<<(N_EDGESC + 255) / 256, 256, 0, stream>>>(dstp, dinv);
  k_dinv<<<(N_VOXELC + 255) / 256, 256, 0, stream>>>(dinv);

  // --- encoder MLP (f16 activations)
  gemm(F0, oWe1, be1, F1, 32, 128, true);
  gemm(F1, oWe2, be2, F0, 128, 128, true);

  // --- GCN layers: conv -> aggregate -> GraphNorm -> ReLU (f16 out)
  auto gcn = [&](const _Float16* x, _Float16* xnext, size_t woff,
                 int K, int N, int nshift, int li) {
    gemm(x, woff, nullptr, G0, K, N, false);
    int qshift = nshift - 2;
    size_t qtot = (size_t)N_VOXELC << qshift;
    k_self_init4<<<(unsigned)((qtot + 255) / 256), 256, 0, stream>>>(G0, dinv, bc[li], G1, qshift);
    size_t eqtot = (size_t)N_EDGESC << qshift;
    k_edge_scatter4<<<(unsigned)((eqtot + 255) / 256), 256, 0, stream>>>(
        srcp, dstp, G0, dinv, G1, qshift);
    k_zero<<<1, 256, 0, stream>>>(stats, 2 * N);
    k_gn_stats<<<(N_VOXELC + GN_CHUNK - 1) / GN_CHUNK, N, 0, stream>>>(G1, stats, N);
    size_t tot = (size_t)N_VOXELC << nshift;
    k_gn_apply<<<(unsigned)((tot + 255) / 256), 256, 0, stream>>>(
        G1, stats, gw[li], gb[li], gm[li], xnext, nshift);
  };
  gcn(F0, F1, oWc0, 128, 64, 6, 0);
  gcn(F1, F0, oWc1, 64, 32, 5, 1);
  gcn(F0, F1, oWc2, 32, 64, 6, 2);
  gcn(F1, F0, oWc3, 64, 128, 7, 3);

  // --- decoder
  gemm(F0, oWd0, bd0, F1, 128, 64, true);
  gemm(F1, oWd1, bd1, F0, 64, 32, true);
  gemm(F0, oWd2, bd2, F1, 32, 16, true);
  k_final<<<(N_VOXELC + 255) / 256, 256, 0, stream>>>(F1, Wd3, bd3, out);

  (void)in_sizes; (void)n_in; (void)out_size; (void)ws_size;
}
